// UniversalNeuralLayer_50964081934812
// MI455X (gfx1250) — matrix operand order repaired and checked by reading, not compiled, not run on hardware
//
#include <hip/hip_runtime.h>
#include <math.h>

#define B_DIM   8192
#define IN_DIM  4096
#define OUT_DIM 2048
#define DEPTH   9

#define BM 128
#define BN 128
#define BK 64
#define PITCH 72   // halfwords per LDS row (64 data + 8 pad) = 144B, 16B-aligned

typedef __attribute__((ext_vector_type(16))) __bf16 v16bf;
typedef __attribute__((ext_vector_type(8)))  __bf16 v8bf;
typedef __attribute__((ext_vector_type(8)))  float  v8f;

// Pack two f32 -> two bf16 (RNE) in one 32-bit lane. No 16-bit register types.
static __device__ __forceinline__ unsigned int pack2(float x, float y) {
#if __has_builtin(__builtin_amdgcn_cvt_pk_bf16_f32)
    auto r = __builtin_amdgcn_cvt_pk_bf16_f32(x, y);
    return __builtin_bit_cast(unsigned int, r);
#else
    unsigned int a = __float_as_uint(x);
    unsigned int b = __float_as_uint(y);
    a = (a + 0x7fffu + ((a >> 16) & 1u)) >> 16;
    b = (b + 0x7fffu + ((b >> 16) & 1u)) & 0xffff0000u;
    return a | b;
#endif
}

// Low 32 bits of a generic pointer to a __shared__ object = LDS byte offset.
static __device__ __forceinline__ unsigned int lds_off(const void* p) {
    return (unsigned int)(unsigned long long)(uintptr_t)p;
}

// One wave-level async global->LDS 16B-per-lane copy (CDNA5 TDM-lite path).
// GVS addressing: global address = 64-bit SGPR base + 32-bit VGPR offset; LDS address from the VDST VGPR.
static __device__ __forceinline__ void async_ld128(unsigned int lds_addr,
                                                   unsigned int goff,
                                                   unsigned long long base) {
    asm volatile("global_load_async_to_lds_b128 %0, %1, %2"
                 :: "v"(lds_addr), "v"(goff), "s"(base) : "memory");
}

// bp[o] = bias[o] + (1/IN_DIM) * prod_{d,c} cos^2(U[d,o,c])
// (reference scan collapses: state stays constant along axis 1; each step
//  multiplies it by cos(angle))
__global__ void universal_bp_kernel(const float* __restrict__ U,
                                    const float* __restrict__ bias,
                                    float* __restrict__ bp) {
    int o = blockIdx.x * blockDim.x + threadIdx.x;
    if (o >= OUT_DIM) return;
    float p = 1.0f;
#pragma unroll
    for (int d = 0; d < DEPTH; ++d) {
        const float* base = U + ((size_t)d * IN_DIM + o) * OUT_DIM;
#pragma unroll
        for (int c = 0; c < 3; ++c) {
            float cs = cosf(base[c]);
            p *= cs * cs;
        }
    }
    bp[o] = bias[o] + p * (1.0f / (float)IN_DIM);
}

// f32 row-major -> bf16 row-major (n4 = element_count/4)
__global__ void cvt_bf16_kernel(const float* __restrict__ in,
                                unsigned int* __restrict__ out, long n4) {
    long i = (long)blockIdx.x * blockDim.x + threadIdx.x;
    long stride = (long)gridDim.x * blockDim.x;
    for (; i < n4; i += stride) {
        float4 v = ((const float4*)in)[i];
        out[2 * i + 0] = pack2(v.x, v.y);
        out[2 * i + 1] = pack2(v.z, v.w);
    }
}

// W (K x N f32, row-major) -> Wt (N x K bf16, row-major) : 32x32 LDS tile transpose
__global__ __launch_bounds__(256)
void cvt_transpose_w_kernel(const float* __restrict__ W,
                            unsigned short* __restrict__ Wt) {
    __shared__ float tile[32][33];
    const int tx = threadIdx.x & 31;
    const int ty = threadIdx.x >> 5;           // 0..7
    const int nb = blockIdx.x * 32;            // N tile base
    const int kb = blockIdx.y * 32;            // K tile base
#pragma unroll
    for (int r = 0; r < 4; ++r)
        tile[ty + r * 8][tx] = W[(size_t)(kb + ty + r * 8) * OUT_DIM + nb + tx];
    __syncthreads();
#pragma unroll
    for (int r = 0; r < 4; ++r) {
        int n = nb + ty + r * 8;
        Wt[(size_t)n * IN_DIM + kb + tx] =
            (unsigned short)(pack2(tile[tx][ty + r * 8], 0.0f) & 0xffffu);
    }
}

// ---- shared compute core: fragments + 16 WMMAs per K-chunk, epilogue ----

__device__ __forceinline__ void wmma_chunk(
    const unsigned short (*Alds)[PITCH], const unsigned short (*Blds)[PITCH],
    int wm, int wn, int half, int l16, v8f acc[2][4]) {
#pragma unroll
    for (int ks = 0; ks < 2; ++ks) {
        const int ko = ks * 32;
        // A 16x32 bf16 per-lane layout (ISA 7.12.2):
        // K runs [8h, 8h+8) and [16+8h, 16+8h+8)
        v16bf a[2];
#pragma unroll
        for (int i = 0; i < 2; ++i) {
            int m = wm + i * 16 + l16;
            v8bf lo = *(const v8bf*)&Alds[m][ko + half * 8];
            v8bf hi = *(const v8bf*)&Alds[m][ko + 16 + half * 8];
#pragma unroll
            for (int e = 0; e < 8; ++e) { a[i][e] = lo[e]; a[i][e + 8] = hi[e]; }
        }
        // B 32x16 bf16: lane column n holds K [8h, 8h+8) and [16+8h, 16+8h+8)
        v16bf b[4];
#pragma unroll
        for (int j = 0; j < 4; ++j) {
            int n = wn + j * 16 + l16;
            v8bf lo = *(const v8bf*)&Blds[n][ko + half * 8];
            v8bf hi = *(const v8bf*)&Blds[n][ko + 16 + half * 8];
#pragma unroll
            for (int e = 0; e < 8; ++e) { b[j][e] = lo[e]; b[j][e + 8] = hi[e]; }
        }
#pragma unroll
        for (int i = 0; i < 2; ++i)
#pragma unroll
            for (int j = 0; j < 4; ++j)
                acc[i][j] = __builtin_amdgcn_wmma_f32_16x16x32_bf16(
                    false, a[i], false, b[j], (short)0, acc[i][j], false, false);
    }
}

__device__ __forceinline__ void epilogue(
    int m0, int n0, int wm, int wn, int half, int l16,
    const float* __restrict__ bp, float* __restrict__ out, v8f acc[2][4]) {
#pragma unroll
    for (int j = 0; j < 4; ++j) {
        const int n   = n0 + wn + j * 16 + l16;
        const float add = bp[n];
#pragma unroll
        for (int i = 0; i < 2; ++i)
#pragma unroll
            for (int r = 0; r < 8; ++r) {
                const int m = m0 + wm + i * 16 + r + 8 * half;
                out[(size_t)m * OUT_DIM + n] = tanhf(acc[i][j][r] + add);
            }
    }
}

// ---- primary path: pre-converted bf16 A [M][K], B [N][K];
//      double-buffered LDS fed by GLOBAL_LOAD_ASYNC_TO_LDS_B128 ----
__global__ __launch_bounds__(256, 1)
void gemm_tanh_pre_kernel(const unsigned short* __restrict__ Xb,
                          const unsigned short* __restrict__ Wt,
                          const float* __restrict__ bp, float* __restrict__ out) {
    __shared__ unsigned short Alds[2][BM][PITCH];
    __shared__ unsigned short Blds[2][BN][PITCH];

    const int m0 = blockIdx.x * BM, n0 = blockIdx.y * BN;
    const int t = threadIdx.x, lane = t & 31, wave = t >> 5;
    const int wm = (wave >> 1) * 32, wn = (wave & 1) * 64;
    const int half = lane >> 4, l16 = lane & 15;

    v8f acc[2][4] = {};
    const int KT = IN_DIM / BK;                     // 64

    // Per-lane invariant addresses: 4 16B chunks each for A and B per K-chunk.
    unsigned int ldsA[4], ldsB[4], gA[4], gB[4];
#pragma unroll
    for (int u = 0; u < 4; ++u) {
        int f = t + u * 256, ar = f >> 3, ac = f & 7;       // 8 groups per row
        ldsA[u] = lds_off(&Alds[0][ar][ac * 8]);
        ldsB[u] = lds_off(&Blds[0][ar][ac * 8]);
        gA[u] = (unsigned int)(((m0 + ar) * IN_DIM + ac * 8) * 2);  // bytes
        gB[u] = (unsigned int)(((n0 + ar) * IN_DIM + ac * 8) * 2);  // bytes
    }
    const unsigned int bufA = (unsigned int)sizeof(Alds[0]);        // 18432 B
    const unsigned int bufB = (unsigned int)sizeof(Blds[0]);

    auto issue = [&](int kt, int buf) {
        unsigned long long sA = (unsigned long long)(uintptr_t)Xb +
                                (unsigned long long)kt * (BK * 2);
        unsigned long long sB = (unsigned long long)(uintptr_t)Wt +
                                (unsigned long long)kt * (BK * 2);
        unsigned int oa = buf ? bufA : 0u;
        unsigned int ob = buf ? bufB : 0u;
#pragma unroll
        for (int u = 0; u < 4; ++u) {
            async_ld128(ldsA[u] + oa, gA[u], sA);
            async_ld128(ldsB[u] + ob, gB[u], sB);
        }
    };

    issue(0, 0);
    issue(1, 1);

    for (int kt = 0; kt < KT; ++kt) {
        const int cur = kt & 1;
        // Async loads complete in order: leave the newest group (8) in flight,
        // require everything older (the current chunk) to have landed.
        if (kt + 1 < KT) {
            asm volatile("s_wait_asynccnt 0x8" ::: "memory");
        } else {
            asm volatile("s_wait_asynccnt 0x0" ::: "memory");
        }
        __syncthreads();                    // LDS writes visible to all waves
        wmma_chunk(Alds[cur], Blds[cur], wm, wn, half, l16, acc);
        __syncthreads();                    // everyone done reading buf[cur]
        if (kt + 2 < KT) issue(kt + 2, cur);
    }
    epilogue(m0, n0, wm, wn, half, l16, bp, out, acc);
}

// ---- fallback path: f32 inputs, convert + transpose while staging (direct) ----
__global__ __launch_bounds__(256, 1)
void gemm_tanh_f32_kernel(const float* __restrict__ X,
                          const float* __restrict__ W,
                          const float* __restrict__ bp, float* __restrict__ out) {
    __shared__ unsigned short Alds[BM][PITCH];
    __shared__ unsigned short Blds[BN][PITCH];

    const int m0 = blockIdx.x * BM, n0 = blockIdx.y * BN;
    const int t = threadIdx.x, lane = t & 31, wave = t >> 5;
    const int wm = (wave >> 1) * 32, wn = (wave & 1) * 64;
    const int half = lane >> 4, l16 = lane & 15;

    v8f acc[2][4] = {};
    const int KT = IN_DIM / BK;

    for (int kt = 0; kt < KT; ++kt) {
        __syncthreads();                    // previous compute done reading LDS
        // A: direct load -> pack -> store (short live ranges)
#pragma unroll
        for (int u = 0; u < 8; ++u) {
            int f = t + u * 256, ar = f >> 4, ac = f & 15;   // 16 float4/row
            float4 v = *(const float4*)(X + (size_t)(m0 + ar) * IN_DIM + kt * BK + ac * 4);
            uint2 q = make_uint2(pack2(v.x, v.y), pack2(v.z, v.w));
            *(uint2*)&Alds[ar][ac * 4] = q;
        }
        // B: wave loads full rows (coalesced); lane gets 4 n at 8 consecutive k
        {
            float bF[8][4];
#pragma unroll
            for (int u = 0; u < 8; ++u) {
                float4 w = *(const float4*)(W + (size_t)(kt * BK + wave * 8 + u) * OUT_DIM +
                                            n0 + lane * 4);
                bF[u][0] = w.x; bF[u][1] = w.y; bF[u][2] = w.z; bF[u][3] = w.w;
            }
#pragma unroll
            for (int e = 0; e < 4; ++e) {
                int n = lane * 4 + e;
                uint4 q;
                q.x = pack2(bF[0][e], bF[1][e]);
                q.y = pack2(bF[2][e], bF[3][e]);
                q.z = pack2(bF[4][e], bF[5][e]);
                q.w = pack2(bF[6][e], bF[7][e]);
                *(uint4*)&Blds[n][wave * 8] = q;   // 16B-aligned (n*144 + wave*16)
            }
        }
        __syncthreads();
        wmma_chunk(Alds, Blds, wm, wn, half, l16, acc);
    }
    epilogue(m0, n0, wm, wn, half, l16, bp, out, acc);
}

extern "C" void kernel_launch(void* const* d_in, const int* in_sizes, int n_in,
                              void* d_out, int out_size, void* d_ws, size_t ws_size,
                              hipStream_t stream) {
    (void)in_sizes; (void)n_in; (void)out_size;
    const float* x    = (const float*)d_in[0];   // (8192, 4096)
    const float* U    = (const float*)d_in[1];   // (9, 4096, 2048)
    const float* W    = (const float*)d_in[2];   // (4096, 2048)
    const float* bias = (const float*)d_in[3];   // (2048,)
    float* out = (float*)d_out;                  // (8192, 2048)

    float* bpw = (float*)d_ws;                   // 2048 f32
    const size_t bpBytes  = (size_t)OUT_DIM * sizeof(float);
    const size_t xbfBytes = (size_t)B_DIM * IN_DIM * 2;   // 64 MiB
    const size_t wtBytes  = (size_t)OUT_DIM * IN_DIM * 2; // 16 MiB

    universal_bp_kernel<<<(OUT_DIM + 255) / 256, 256, 0, stream>>>(U, bias, bpw);

    dim3 grid(B_DIM / BM, OUT_DIM / BN);         // 64 x 16 workgroups

    if (ws_size >= bpBytes + xbfBytes + wtBytes) {
        unsigned short* Xbf = (unsigned short*)((char*)d_ws + bpBytes);
        unsigned short* Wt  = (unsigned short*)((char*)d_ws + bpBytes + xbfBytes);
        cvt_bf16_kernel<<<4096, 256, 0, stream>>>(
            x, (unsigned int*)Xbf, (long)B_DIM * IN_DIM / 4);
        dim3 tg(OUT_DIM / 32, IN_DIM / 32);
        cvt_transpose_w_kernel<<<tg, 256, 0, stream>>>(W, Wt);
        gemm_tanh_pre_kernel<<<grid, 256, 0, stream>>>(Xbf, Wt, bpw, out);
    } else {
        gemm_tanh_f32_kernel<<<grid, 256, 0, stream>>>(x, W, bpw, out);
    }
}
